// GemmaCausalAtt_29575144800787
// MI455X (gfx1250) — matrix operand order repaired and checked by reading, not compiled, not run on hardware
//
#include <hip/hip_runtime.h>
#include <hip/hip_bf16.h>

typedef unsigned short ushort_t;
typedef unsigned int uint_t;
typedef __attribute__((ext_vector_type(16))) __bf16 v16bf;
typedef __attribute__((ext_vector_type(8)))  float  v8f;
typedef unsigned int v4u __attribute__((ext_vector_type(4)));
typedef int v8i __attribute__((ext_vector_type(8)));
typedef int v4i __attribute__((ext_vector_type(4)));

#define B_  2
#define T_  2048
#define C_  2048
#define NH_ 16
#define NKV_ 8
#define HD_ 256

union Frag { v16bf v; uint_t u[8]; };

__device__ __forceinline__ ushort_t f2bf(float f) {
    uint_t x = __float_as_uint(f);
    uint_t r = x + 0x7FFFu + ((x >> 16) & 1u);   // round-to-nearest-even
    return (ushort_t)(r >> 16);
}
__device__ __forceinline__ float bf2f(ushort_t h) {
    return __uint_as_float(((uint_t)h) << 16);
}

// ---------------------------------------------------------------------------
// Elementwise converters
// ---------------------------------------------------------------------------
__global__ void k_f32_to_bf16(const float* __restrict__ in, ushort_t* __restrict__ out, size_t n) {
    size_t i = (size_t)blockIdx.x * blockDim.x + threadIdx.x;
    if (i < n) out[i] = f2bf(in[i]);
}

// in: K x N (row-major f32)  ->  out: N x K (row-major bf16)
__global__ void k_conv_transpose(const float* __restrict__ in, ushort_t* __restrict__ out, int K, int N) {
    size_t i = (size_t)blockIdx.x * blockDim.x + threadIdx.x;
    if (i >= (size_t)K * N) return;
    int n = (int)(i % N);
    int k = (int)(i / N);
    out[(size_t)n * K + k] = f2bf(in[i]);
}

// Vb [B,T,NKV,HD] bf16 -> Vt [B,NKV,HD,T] bf16
__global__ void k_transpose_v(const ushort_t* __restrict__ Vb, ushort_t* __restrict__ Vt) {
    size_t i = (size_t)blockIdx.x * blockDim.x + threadIdx.x;
    if (i >= (size_t)B_ * T_ * NKV_ * HD_) return;
    int d = (int)(i % HD_);
    size_t r = i / HD_;
    int h = (int)(r % NKV_);
    size_t r2 = r / NKV_;
    int t = (int)(r2 % T_);
    int b = (int)(r2 / T_);
    Vt[(((size_t)(b * NKV_ + h)) * HD_ + d) * T_ + t] = Vb[i];
}

// In-place RoPE on [B,T,H,256] bf16; thread handles pair (d, d+128)
__global__ void k_rope(ushort_t* __restrict__ X, int H) {
    size_t i = (size_t)blockIdx.x * blockDim.x + threadIdx.x;
    size_t total = (size_t)B_ * T_ * H * (HD_ / 2);
    if (i >= total) return;
    int d = (int)(i % (HD_ / 2));
    size_t r1 = i / (HD_ / 2);                 // (b*T + t)*H + h
    size_t r2 = r1 / H;
    int t = (int)(r2 % T_);
    float ang = (float)t * __expf(-(float)d * (9.21034037f / 128.0f)); // 1/10000^(d/128)
    float c = __cosf(ang), s = __sinf(ang);
    size_t base = r1 * HD_;
    float x1 = bf2f(X[base + d]);
    float x2 = bf2f(X[base + d + 128]);
    X[base + d]       = f2bf(x1 * c - x2 * s);
    X[base + d + 128] = f2bf(x2 * c + x1 * s);
}

// ---------------------------------------------------------------------------
// Tensor Data Mover: DMA a 2D bf16 tile (128 rows x 32 cols) global -> LDS.
// D# per CDNA5 ISA:
//  g0: [1:0]=count=1, [63:32]=lds_addr, [120:64]=global_addr, [127:126]=type=2
//  g1: [17:16]=data_size(1->2B), [79:48]=tensor_dim0, [111:80]=tensor_dim1,
//      [127:112]=tile_dim0, [143:128]=tile_dim1, [207:160]=tensor_dim0_stride
// This toolchain exposes the 6-arg builtin (extra int32x8 group before cpol).
// ---------------------------------------------------------------------------
__device__ __forceinline__ void tdm_load_tile_128x32(const ushort_t* gptr,
                                                     uint_t lds_off, int K) {
    unsigned long long ga = (unsigned long long)(uintptr_t)gptr;
    v4u g0 = { 1u,                                   // count = 1 valid descriptor
               lds_off,                              // LDS byte address
               (uint_t)ga,                           // global_addr[31:0]
               ((uint_t)(ga >> 32) & 0x01FFFFFFu) | (2u << 30) }; // [56:32] + type=2
    uint_t k = (uint_t)K;
    v8i g1 = { (int)(1u << 16),                      // wg_mask=0, data_size=1 (2B)
               (int)(k << 16),                       // tensor_dim0[15:0] in [63:48]
               (int)((128u << 16) | (k >> 16)),      // dim0[31:16], tensor_dim1=128
               (int)(32u << 16),                     // dim1[31:16]=0, tile_dim0=32
               (int)128,                             // tile_dim1=128, tile_dim2=0
               (int)k,                               // tensor_dim0_stride[31:0] = K
               0, 0 };                               // stride hi, dim1_stride = 0
    v4i gz4 = { 0, 0, 0, 0 };
    v8i gz8 = { 0, 0, 0, 0, 0, 0, 0, 0 };
    __builtin_amdgcn_tensor_load_to_lds(g0, g1, gz4, gz4, gz8, 0);
}

// ---------------------------------------------------------------------------
// bf16 GEMM:  C[M,N] = A[M,K] * Bt[N,K]^T   (A, Bt row-major bf16, f32 accum)
// block = 256 thr = 8 waves; block tile 128x128; wave tile 32x64 (2x4 WMMA).
// Tiles staged to LDS by the Tensor Data Mover (wave 0 issues descriptors),
// double-buffered on TENSORcnt.
// ---------------------------------------------------------------------------
template <int WRITE_F32>
__global__ __launch_bounds__(256)
void k_gemm_bf16(const ushort_t* __restrict__ A, const ushort_t* __restrict__ Bt,
                 void* __restrict__ Cout, int M, int N, int K) {
    (void)M;
    __shared__ ushort_t lA[2][128 * 32];
    __shared__ ushort_t lB[2][128 * 32];
    const int tid   = threadIdx.x;
    const int wid   = tid >> 5;
    const int lane  = tid & 31;
    const int l16   = lane & 15;
    const int lhalf = lane >> 4;
    const int wm    = wid >> 1;        // 0..3 -> M offset 32*wm
    const int wn    = wid & 1;         // 0..1 -> N offset 64*wn
    const int bm    = blockIdx.y * 128;
    const int bn    = blockIdx.x * 128;

    v8f acc[2][4];
#pragma unroll
    for (int i = 0; i < 2; i++)
#pragma unroll
        for (int j = 0; j < 4; j++) acc[i][j] = (v8f){};

    const ushort_t* Abase = A  + (size_t)bm * K;
    const ushort_t* Bbase = Bt + (size_t)bn * K;
    const uint_t ldsA[2] = { (uint_t)(uintptr_t)&lA[0][0], (uint_t)(uintptr_t)&lA[1][0] };
    const uint_t ldsB[2] = { (uint_t)(uintptr_t)&lB[0][0], (uint_t)(uintptr_t)&lB[1][0] };

    const int nk = K >> 5;
    if (wid == 0) {                                   // prime buffer 0
        tdm_load_tile_128x32(Abase, ldsA[0], K);
        tdm_load_tile_128x32(Bbase, ldsB[0], K);
    }

    for (int i = 0; i < nk; i++) {
        const int buf = i & 1;
        if (wid == 0) __builtin_amdgcn_s_wait_tensorcnt(0);  // tile i landed in LDS
        __syncthreads();                                      // visible to all waves
        if (wid == 0 && (i + 1) < nk) {                       // prefetch tile i+1
            tdm_load_tile_128x32(Abase + (size_t)(i + 1) * 32, ldsA[buf ^ 1], K);
            tdm_load_tile_128x32(Bbase + (size_t)(i + 1) * 32, ldsB[buf ^ 1], K);
        }

        Frag a[2], b[4];
#pragma unroll
        for (int mt = 0; mt < 2; mt++) {
            const uint_t* p = (const uint_t*)(&lA[buf][(wm * 32 + mt * 16 + l16) * 32]);
#pragma unroll
            for (int v = 0; v < 8; v++) {
                int k = lhalf * 16 + 2 * v;
                a[mt].u[v] = p[k >> 1];
            }
        }
#pragma unroll
        for (int nt = 0; nt < 4; nt++) {
            const uint_t* p = (const uint_t*)(&lB[buf][(wn * 64 + nt * 16 + l16) * 32]);
#pragma unroll
            for (int v = 0; v < 8; v++) {
                int k = lhalf * 16 + 2 * v;
                b[nt].u[v] = p[k >> 1];
            }
        }
#pragma unroll
        for (int mt = 0; mt < 2; mt++)
#pragma unroll
            for (int nt = 0; nt < 4; nt++)
                acc[mt][nt] = __builtin_amdgcn_wmma_f32_16x16x32_bf16(
                    false, a[mt].v, false, b[nt].v, (short)0, acc[mt][nt], false, false);
        __syncthreads();   // reads of buf done before it is DMA-overwritten (i+2)
    }

#pragma unroll
    for (int mt = 0; mt < 2; mt++)
#pragma unroll
        for (int nt = 0; nt < 4; nt++) {
            int col = bn + wn * 64 + nt * 16 + l16;
#pragma unroll
            for (int r = 0; r < 8; r++) {
                int rrow = bm + wm * 32 + mt * 16 + r + 8 * lhalf;
                float val = acc[mt][nt][r];
                if (WRITE_F32) ((float*)Cout)[(size_t)rrow * N + col] = val;
                else           ((ushort_t*)Cout)[(size_t)rrow * N + col] = f2bf(val);
            }
        }
}

// ---------------------------------------------------------------------------
// Flash attention, softcap + causal, bf16 WMMA, f32 online softmax.
// grid (T/64, NH, B); block 128 thr = 4 waves; each wave owns 16 q rows.
// Q,K: [B,T,H,256] bf16 (rope'd); Vt: [B,NKV,256,T] bf16; Y: [B,T,NH,256] bf16
// ---------------------------------------------------------------------------
__global__ __launch_bounds__(128)
void k_attn(const ushort_t* __restrict__ Q, const ushort_t* __restrict__ Km,
            const ushort_t* __restrict__ Vt, ushort_t* __restrict__ Y) {
    __shared__ ushort_t pstage[4][16 * 32];
    const int lane  = threadIdx.x & 31;
    const int wid   = threadIdx.x >> 5;
    const int l16   = lane & 15;
    const int lhalf = lane >> 4;
    const int h  = blockIdx.y;
    const int b  = blockIdx.z;
    const int hk = h >> 1;                      // NH/NKV = 2
    const int qbase = blockIdx.x * 64 + wid * 16;

    v8f O[16];
#pragma unroll
    for (int i = 0; i < 16; i++) O[i] = (v8f){};
    float m[8], l[8];
#pragma unroll
    for (int r = 0; r < 8; r++) { m[r] = -3.0e38f; l[r] = 0.f; }

    const float scale = 0.0625f;                // 1/sqrt(256)
    const int kend = qbase + 15;

    for (int kt0 = 0; kt0 <= kend; kt0 += 32) {
#pragma unroll
        for (int j = 0; j < 2; j++) {
            const int key = kt0 + j * 16 + l16;
            v8f S = (v8f){};
#pragma unroll
            for (int c = 0; c < 8; c++) {       // d chunks of 32
                Frag qa, kb;
                {   // Q A-fragment: lane row = qbase + l16
                    const uint_t* p = (const uint_t*)(Q +
                        (((size_t)(b * T_ + (qbase + l16))) * NH_ + h) * HD_);
#pragma unroll
                    for (int v = 0; v < 8; v++) {
                        int k = c * 32 + lhalf * 16 + 2 * v;
                        qa.u[v] = p[k >> 1];
                    }
                }
                {   // K^T B-fragment: lane col = key
                    const uint_t* p = (const uint_t*)(Km +
                        (((size_t)(b * T_ + key)) * NKV_ + hk) * HD_);
#pragma unroll
                    for (int v = 0; v < 8; v++) {
                        int k = c * 32 + lhalf * 16 + 2 * v;
                        kb.u[v] = p[k >> 1];
                    }
                }
                S = __builtin_amdgcn_wmma_f32_16x16x32_bf16(
                        false, qa.v, false, kb.v, (short)0, S, false, false);
            }
            // softcap + causal mask + online softmax update
#pragma unroll
            for (int r = 0; r < 8; r++) {
                int qrow = qbase + r + 8 * lhalf;
                float s = S[r] * scale;
                s = 50.f * tanhf(s * 0.02f);
                if (key > qrow) s = -3.0e38f;

                float rmax = s;
#pragma unroll
                for (int off = 8; off; off >>= 1)
                    rmax = fmaxf(rmax, __shfl_xor(rmax, off, 32));
                float mn = fmaxf(m[r], rmax);
                float alpha = __expf(m[r] - mn);
                float p = __expf(s - mn);
                float rsum = p;
#pragma unroll
                for (int off = 8; off; off >>= 1)
                    rsum += __shfl_xor(rsum, off, 32);
                l[r] = l[r] * alpha + rsum;
                m[r] = mn;
#pragma unroll
                for (int dt = 0; dt < 16; dt++) O[dt][r] = O[dt][r] * alpha;
                pstage[wid][(r + 8 * lhalf) * 32 + j * 16 + l16] = f2bf(p);
            }
        }
        asm volatile("s_wait_dscnt 0" ::: "memory");   // same-wave DS RAW fence

        Frag pf;   // P as A-fragment (16x32)
        {
            const uint_t* p = (const uint_t*)(&pstage[wid][l16 * 32]);
#pragma unroll
            for (int v = 0; v < 8; v++) {
                int k = lhalf * 16 + 2 * v;
                pf.u[v] = p[k >> 1];
            }
        }
#pragma unroll
        for (int dt = 0; dt < 16; dt++) {       // O += P * V over 32 keys
            Frag vb;
            int d = dt * 16 + l16;
            const uint_t* p = (const uint_t*)(Vt +
                (((size_t)(b * NKV_ + hk)) * HD_ + d) * T_ + kt0);
#pragma unroll
            for (int v = 0; v < 8; v++) {
                int k = lhalf * 16 + 2 * v;
                vb.u[v] = p[k >> 1];
            }
            O[dt] = __builtin_amdgcn_wmma_f32_16x16x32_bf16(
                        false, pf.v, false, vb.v, (short)0, O[dt], false, false);
        }
    }

#pragma unroll
    for (int dt = 0; dt < 16; dt++) {
        int d = dt * 16 + l16;
#pragma unroll
        for (int r = 0; r < 8; r++) {
            int q = qbase + r + 8 * lhalf;
            float val = O[dt][r] / l[r];
            Y[(((size_t)(b * T_ + q)) * NH_ + h) * HD_ + d] = f2bf(val);
        }
    }
}

// ---------------------------------------------------------------------------
extern "C" void kernel_launch(void* const* d_in, const int* in_sizes, int n_in,
                              void* d_out, int out_size, void* d_ws, size_t ws_size,
                              hipStream_t stream) {
    (void)in_sizes; (void)n_in; (void)out_size; (void)ws_size;
    const float* X  = (const float*)d_in[0];
    const float* Wq = (const float*)d_in[1];
    const float* Wk = (const float*)d_in[2];
    const float* Wv = (const float*)d_in[3];
    const float* Wo = (const float*)d_in[4];
    float* out = (float*)d_out;

    // workspace layout (bf16 element offsets)
    ushort_t* ws = (ushort_t*)d_ws;
    ushort_t* Xb  = ws;                      // 8M   (B*T*C)
    ushort_t* Wqt = Xb  + 8388608;           // 8M   (4096x2048 transposed)
    ushort_t* Wkt = Wqt + 8388608;           // 4M
    ushort_t* Wvt = Wkt + 4194304;           // 4M
    ushort_t* Wot = Wvt + 4194304;           // 8M   (2048x4096 transposed)
    ushort_t* Qb  = Wot + 8388608;           // 16M  (B,T,NH,HD)
    ushort_t* Kb  = Qb  + 16777216;          // 8M   (B,T,NKV,HD)
    ushort_t* Vb  = Kb  + 8388608;           // 8M
    ushort_t* Vt  = Vb  + 8388608;           // 8M   (B,NKV,HD,T)
    ushort_t* Yb  = Vt  + 8388608;           // 16M  (B,T,NH,HD)

    const int TPB = 256;
    k_f32_to_bf16<<<(8388608 + TPB - 1) / TPB, TPB, 0, stream>>>(X, Xb, 8388608);
    k_conv_transpose<<<(8388608 + TPB - 1) / TPB, TPB, 0, stream>>>(Wq, Wqt, 2048, 4096);
    k_conv_transpose<<<(4194304 + TPB - 1) / TPB, TPB, 0, stream>>>(Wk, Wkt, 2048, 2048);
    k_conv_transpose<<<(4194304 + TPB - 1) / TPB, TPB, 0, stream>>>(Wv, Wvt, 2048, 2048);
    k_conv_transpose<<<(8388608 + TPB - 1) / TPB, TPB, 0, stream>>>(Wo, Wot, 4096, 2048);

    // QKV projections (bf16 out, f32 accum on WMMA, TDM-staged tiles)
    k_gemm_bf16<0><<<dim3(32, 32), 256, 0, stream>>>(Xb, Wqt, Qb, 4096, 4096, 2048);
    k_gemm_bf16<0><<<dim3(16, 32), 256, 0, stream>>>(Xb, Wkt, Kb, 4096, 2048, 2048);
    k_gemm_bf16<0><<<dim3(16, 32), 256, 0, stream>>>(Xb, Wvt, Vb, 4096, 2048, 2048);

    // RoPE (in place) + V transpose for contiguous PV B-fragments
    k_rope<<<((size_t)B_ * T_ * NH_  * 128 + TPB - 1) / TPB, TPB, 0, stream>>>(Qb, NH_);
    k_rope<<<((size_t)B_ * T_ * NKV_ * 128 + TPB - 1) / TPB, TPB, 0, stream>>>(Kb, NKV_);
    k_transpose_v<<<(8388608 + TPB - 1) / TPB, TPB, 0, stream>>>(Vb, Vt);

    // attention
    k_attn<<<dim3(T_ / 64, NH_, B_), 128, 0, stream>>>(Qb, Kb, Vt, Yb);

    // output projection (f32 out)
    k_gemm_bf16<1><<<dim3(16, 32), 256, 0, stream>>>(Yb, Wot, out, 4096, 2048, 4096);
}
